// ManualBiLSTMLM_35923106464038
// MI455X (gfx1250) — matrix operand order repaired and checked by reading, not compiled, not run on hardware
//
#include <hip/hip_runtime.h>
#include <hip/hip_bf16.h>

// ---------------------------------------------------------------------------
// CDNA5 (gfx1250) BiLSTM LM. wave32. All GEMMs via v_wmma_f32_16x16x32_bf16.
// Weights pre-packed into WMMA B-fragment layout -> inner loop is pure b128
// loads + 4 unconditional WMMAs. Wave id is readfirstlane'd so tile guards are
// scalar branches (EXEC stays all-ones around every WMMA, per ISA §7.12).
// B-operand mode is a template parameter so the k-loop is branch-free.
// ---------------------------------------------------------------------------

typedef __bf16 bf16;
typedef __attribute__((ext_vector_type(16))) __bf16 v16bf;
typedef __attribute__((ext_vector_type(8)))  float  v8f;

#define WMMA_BF16(a, b, c) \
  __builtin_amdgcn_wmma_f32_16x16x32_bf16(false, (a), false, (b), (short)0, (c), false, false)

constexpr int Vv = 32000, Ee = 512, Hh = 512, Bb = 8, Ss = 512;
constexpr int BS = Bb * Ss;                       // 4096 rows everywhere
constexpr float NEGF = -3.4028234663852886e38f;   // finfo(f32).min

// ---- WMMA fragment helpers (contiguous k order: k = 16*(lane/16) + e in A and B) -----

// A (M=16,K=32) bf16, row-major source. lane 0-15: row=lane, K=0..15;
// lane 16-31: row=lane-16, K=16..31.  One contiguous 32B load/lane.
__device__ __forceinline__ v16bf load_a_frag(const bf16* A, int lda, int row0,
                                             int k0, int lane) {
  int r  = row0 + (lane & 15);
  int kb = k0 + ((lane & 16) ? 16 : 0);
  const bf16* p = A + (size_t)r * lda + kb;
  v16bf a;
#pragma unroll
  for (int e = 0; e < 16; ++e) a[e] = p[e];
  return a;
}

// B fragment from a row-major (N x K) matrix (i.e. multiply by its transpose):
// lane 0-15 hold K=kb..kb+15 of row n -> fully contiguous 32B load per lane.
__device__ __forceinline__ v16bf load_bt_frag(const bf16* Bt, int ldbt, int k0,
                                              int col0, int lane) {
  int n  = col0 + (lane & 15);
  int kb = k0 + ((lane & 16) ? 16 : 0);
  const bf16* p = Bt + (size_t)n * ldbt + kb;
  v16bf b;
#pragma unroll
  for (int e = 0; e < 16; ++e) b[e] = p[e];
  return b;
}

// Packed-B layout: fragment f = nt*Ktot + kt holds the (32k x 16n) tile; the
// 32 lanes' 16 bf16 values are stored contiguously: offset ((f*32+lane)*16).
// -> per-lane fragment load is one aligned 32B access (SGPR base + lane*32).
__device__ __forceinline__ v16bf load_bp_frag(const bf16* P, size_t frag, int lane) {
  return ((const v16bf*)P)[frag * 32 + lane];
}

// ---- weight packing: f32 (K x N) row-major -> packed bf16 fragments --------
__global__ void pack_w_kernel(const float* __restrict__ W, int K, int N,
                              bf16* __restrict__ P) {
  long long i = (long long)blockIdx.x * 256 + threadIdx.x;  // over K*N
  if (i >= (long long)K * N) return;
  int Ktot = K >> 5;
  int e = (int)(i & 15);
  long long fi = i >> 4;
  int lane = (int)(fi & 31);
  long long f = fi >> 5;
  int kt = (int)(f % Ktot);
  int nt = (int)(f / Ktot);
  int k = kt * 32 + ((lane & 16) ? 16 : 0) + e;
  int n = nt * 16 + (lane & 15);
  P[i] = (bf16)W[(size_t)k * N + n];
}

// Same but bf16 source, batched (used for enc -> context GEMM B operand).
__global__ void pack_b_bf16_kernel(const bf16* __restrict__ B, long long sB,
                                   int K, int N, bf16* __restrict__ P,
                                   long long sP) {
  long long i = (long long)blockIdx.x * 256 + threadIdx.x;  // over K*N
  if (i >= (long long)K * N) return;
  B += (size_t)blockIdx.z * sB;
  P += (size_t)blockIdx.z * sP;
  int Ktot = K >> 5;
  int e = (int)(i & 15);
  long long fi = i >> 4;
  int lane = (int)(fi & 31);
  long long f = fi >> 5;
  int kt = (int)(f % Ktot);
  int nt = (int)(f / Ktot);
  int k = kt * 32 + ((lane & 16) ? 16 : 0) + e;
  int n = nt * 16 + (lane & 15);
  P[i] = B[(size_t)k * N + n];
}

// ---- generic batched bf16 GEMM, f32 accumulate, fused epilogue -------------
// C[z] = act( A[z] @ B[z] * scale + bias ) + residual ; writes f32 and/or bf16.
// Each wave computes a 16(M) x 64(N) strip; requires (N/16) % 4 == 0 (true for
// all shapes here), so an active wave always owns 4 full tiles: the k-loop is
// branch-free. BTRANS=0: B is packed fragments (ldbK=Ktot, kt0=first k-tile).
// BTRANS=1: B is a row-major (N x K) matrix (ldbK=row pitch).
template <int BTRANS>
__global__ void gemm_bf16_kernel(
    const bf16* __restrict__ A, int lda, long long sA,
    const bf16* __restrict__ B, int ldbK, long long sB, int kt0,
    const float* __restrict__ bias,
    float* __restrict__ Cf, bf16* __restrict__ Cb, int ldc, long long sC,
    const float* __restrict__ res, int ldr, long long sR,
    int N, int K, float scale, int actk) {
  int z = blockIdx.z;
  A += (size_t)z * sA;
  B += (size_t)z * sB;
  if (Cf)  Cf  += (size_t)z * sC;
  if (Cb)  Cb  += (size_t)z * sC;
  if (res) res += (size_t)z * sR;

  int lane = threadIdx.x & 31;
  int wv = __builtin_amdgcn_readfirstlane(threadIdx.x >> 5);  // SGPR wave id
  int nt0 = (blockIdx.x * 8 + wv) * 4;
  if (nt0 * 16 >= N) return;  // scalar branch; EXEC untouched
  int mrow = blockIdx.y * 16;
  int nkt = K >> 5;

  v8f acc0 = {}, acc1 = {}, acc2 = {}, acc3 = {};
#pragma unroll 2
  for (int kk = 0; kk < nkt; ++kk) {
    v16bf a = load_a_frag(A, lda, mrow, kk * 32, lane);
    v16bf b0, b1, b2, b3;
    if constexpr (BTRANS) {
      b0 = load_bt_frag(B, ldbK, kk * 32, (nt0 + 0) * 16, lane);
      b1 = load_bt_frag(B, ldbK, kk * 32, (nt0 + 1) * 16, lane);
      b2 = load_bt_frag(B, ldbK, kk * 32, (nt0 + 2) * 16, lane);
      b3 = load_bt_frag(B, ldbK, kk * 32, (nt0 + 3) * 16, lane);
    } else {
      size_t f = (size_t)nt0 * ldbK + kt0 + kk;
      if (kk + 4 < nkt)  // stream packed B ahead (global_prefetch_b8)
        __builtin_prefetch((const void*)(B + (f + 4) * 32 * 16), 0, 1);
      b0 = load_bp_frag(B, f, lane);
      b1 = load_bp_frag(B, f + (size_t)ldbK, lane);
      b2 = load_bp_frag(B, f + (size_t)2 * ldbK, lane);
      b3 = load_bp_frag(B, f + (size_t)3 * ldbK, lane);
    }
    acc0 = WMMA_BF16(a, b0, acc0);
    acc1 = WMMA_BF16(a, b1, acc1);
    acc2 = WMMA_BF16(a, b2, acc2);
    acc3 = WMMA_BF16(a, b3, acc3);
  }

  int rbase = mrow + ((lane & 16) ? 8 : 0);
  auto epilogue = [&](int i, const v8f& acc) {
    int col = (nt0 + i) * 16 + (lane & 15);
    float bcol = bias ? bias[col] : 0.0f;
#pragma unroll
    for (int r = 0; r < 8; ++r) {
      float v = acc[r] * scale + bcol;
      if (actk == 1)      v = tanhf(v);
      else if (actk == 2) v = 0.5f * v * (1.0f + erff(v * 0.70710678118654752f));
      int row = rbase + r;
      if (res) v += res[(size_t)row * ldr + col];
      if (Cf)  Cf[(size_t)row * ldc + col] = v;
      if (Cb)  Cb[(size_t)row * ldc + col] = (bf16)v;
    }
  };
  epilogue(0, acc0);
  epilogue(1, acc1);
  epilogue(2, acc2);
  epilogue(3, acc3);
}

// ---- persistent BiLSTM recurrence ------------------------------------------
// One block per direction (blockIdx.x: 0=fwd, 1=bwd), 1024 threads = 32 waves.
// h/c state + bf16 A-operand live in LDS; per step one 16x2048x512 WMMA GEMM
// (gates = h @ Wh) with Wh streamed as packed fragments, added to the
// precomputed x @ Wx + bias (gpre).
__global__ void lstm_rec_kernel(
    const bf16* __restrict__ WfP, const bf16* __restrict__ WbP,
    int Ktot, int kt0,
    const float* __restrict__ gpre_f, const float* __restrict__ gpre_b,
    const float* __restrict__ h0f, const float* __restrict__ c0f,
    const float* __restrict__ h0b, const float* __restrict__ c0b,
    const int* __restrict__ mask,
    bf16* __restrict__ outB, float* __restrict__ outF, int ldo) {
  int dir = blockIdx.x;
  const bf16*  WP   = dir ? WbP    : WfP;
  const float* gpre = dir ? gpre_b : gpre_f;
  const float* h0   = dir ? h0b    : h0f;
  const float* c0   = dir ? c0b    : c0f;
  int outoff = dir ? Hh : 0;

  extern __shared__ char smem[];
  bf16*  h_bf = (bf16*)smem;                               // 16 x 512 (rows 8..15 = 0)
  float* hF   = (float*)(smem + 16 * 512 * sizeof(bf16));  // 8 x 512
  float* cF   = hF + 8 * 512;                              // 8 x 512
  float* gL   = cF + 8 * 512;                              // 8 x 2048 gate staging

  int tid = threadIdx.x;
  for (int idx = tid; idx < 16 * 512; idx += 1024) {
    int b = idx >> 9, j = idx & 511;
    h_bf[idx] = (b < 8) ? (bf16)h0[j] : (bf16)0.0f;
  }
  for (int idx = tid; idx < 8 * 512; idx += 1024) {
    int j = idx & 511;
    hF[idx] = h0[j];
    cF[idx] = c0[j];
  }

  int wv = __builtin_amdgcn_readfirstlane(tid >> 5);  // SGPR wave id
  int lane = tid & 31;
  int nt0 = wv * 4;  // this wave's 4 N-tiles (128 tiles over 32 waves)
  for (int ti = 0; ti < Ss; ++ti) {
    int t = dir ? (Ss - 1 - ti) : ti;
    __syncthreads();  // h_bf from previous step is complete

    // gates[16,2048] = h_bf[16,512] @ Wh[512,2048]
    v8f acc0 = {}, acc1 = {}, acc2 = {}, acc3 = {};
#pragma unroll 1
    for (int kk = 0; kk < 16; ++kk) {
      v16bf a = load_a_frag(h_bf, 512, 0, kk * 32, lane);
      size_t f = (size_t)nt0 * Ktot + kt0 + kk;
      v16bf b0 = load_bp_frag(WP, f, lane);
      v16bf b1 = load_bp_frag(WP, f + (size_t)Ktot, lane);
      v16bf b2 = load_bp_frag(WP, f + (size_t)2 * Ktot, lane);
      v16bf b3 = load_bp_frag(WP, f + (size_t)3 * Ktot, lane);
      acc0 = WMMA_BF16(a, b0, acc0);
      acc1 = WMMA_BF16(a, b1, acc1);
      acc2 = WMMA_BF16(a, b2, acc2);
      acc3 = WMMA_BF16(a, b3, acc3);
    }
    if (lane < 16) {  // only batch rows 0..7 are real; after all WMMAs
      const v8f* accs[4] = {&acc0, &acc1, &acc2, &acc3};
#pragma unroll
      for (int i = 0; i < 4; ++i) {
        int col = (nt0 + i) * 16 + lane;
#pragma unroll
        for (int r = 0; r < 8; ++r) gL[r * 2048 + col] = (*accs[i])[r];
      }
    }
    __syncthreads();

    // elementwise LSTM cell: 8 batches x 512 hidden over 1024 threads
#pragma unroll 1
    for (int i = 0; i < 4; ++i) {
      int idx = i * 1024 + tid;
      int b = idx >> 9, j = idx & 511;
      size_t orow = (size_t)b * Ss + t;
      const float* gp = gpre + orow * 2048;
      float ig = gL[b * 2048 + j]        + gp[j];
      float fg = gL[b * 2048 + 512 + j]  + gp[512 + j];
      float og = gL[b * 2048 + 1024 + j] + gp[1024 + j];
      float gg = gL[b * 2048 + 1536 + j] + gp[1536 + j];
      float c_old = cF[b * 512 + j];
      float si = 1.0f / (1.0f + expf(-ig));
      float sf = 1.0f / (1.0f + expf(-fg));
      float so = 1.0f / (1.0f + expf(-og));
      float c2 = sf * c_old + si * tanhf(gg);
      float h2 = so * tanhf(c2);
      int am = mask[orow];
      float hn = am ? h2 : hF[b * 512 + j];
      float cn = am ? c2 : c_old;
      hF[b * 512 + j] = hn;
      cF[b * 512 + j] = cn;
      h_bf[b * 512 + j] = (bf16)hn;
      outB[orow * ldo + outoff + j] = am ? (bf16)h2 : (bf16)0.0f;
      if (outF) outF[orow * ldo + outoff + j] = am ? h2 : 0.0f;
    }
  }
}

// ---- embedding gather + pad mask ------------------------------------------
__global__ void embed_kernel(const int* __restrict__ x, const float* __restrict__ emb,
                             bf16* __restrict__ X0, int* __restrict__ mask) {
  int row = blockIdx.x;  // b*S + s
  int tok = x[row];
  if (threadIdx.x == 0) mask[row] = (tok != 0);
  const float* e = emb + (size_t)tok * Ee;  // emb[0] == 0 handles padding
  bf16* o = X0 + (size_t)row * Ee;
  for (int c = threadIdx.x; c < Ee; c += 256) o[c] = (bf16)e[c];
}

// ---- masked softmax over keys; zero rows for invalid queries ---------------
__global__ void attn_softmax_kernel(const float* __restrict__ scores,
                                    const int* __restrict__ mask,
                                    bf16* __restrict__ attn) {
  int row = blockIdx.x;  // b*S + q
  int b = row >> 9, q = row & 511;
  const float* sr = scores + (size_t)row * Ss;
  bf16* ar = attn + (size_t)row * Ss;
  int qok = mask[b * Ss + q];
  __shared__ float red[256];
  __shared__ float sv[512];
  int t = threadIdx.x;
  float mx = NEGF;
  for (int k = t; k < Ss; k += 256) {
    float v = mask[b * Ss + k] ? sr[k] : NEGF;
    sv[k] = v;
    mx = fmaxf(mx, v);
  }
  red[t] = mx; __syncthreads();
  for (int off = 128; off > 0; off >>= 1) {
    if (t < off) red[t] = fmaxf(red[t], red[t + off]);
    __syncthreads();
  }
  mx = red[0]; __syncthreads();
  float s = 0.0f;
  for (int k = t; k < Ss; k += 256) { float e = expf(sv[k] - mx); sv[k] = e; s += e; }
  red[t] = s; __syncthreads();
  for (int off = 128; off > 0; off >>= 1) {
    if (t < off) red[t] += red[t + off];
    __syncthreads();
  }
  float inv = 1.0f / red[0];
  for (int k = t; k < Ss; k += 256)
    ar[k] = qok ? (bf16)(sv[k] * inv) : (bf16)0.0f;
}

// ---- layernorm (population var), optional per-row zero mask, bf16 out ------
__global__ void layernorm_kernel(const float* __restrict__ X, int n,
                                 const float* __restrict__ g, const float* __restrict__ bb,
                                 const int* __restrict__ mask, float eps,
                                 bf16* __restrict__ outB) {
  int row = blockIdx.x;
  const float* xr = X + (size_t)row * n;
  bf16* orow = outB + (size_t)row * n;
  int t = threadIdx.x;
  __shared__ float r1[256], r2[256];
  float s = 0.0f, s2 = 0.0f;
  for (int j = t; j < n; j += 256) { float v = xr[j]; s += v; s2 += v * v; }
  r1[t] = s; r2[t] = s2; __syncthreads();
  for (int off = 128; off > 0; off >>= 1) {
    if (t < off) { r1[t] += r1[t + off]; r2[t] += r2[t + off]; }
    __syncthreads();
  }
  float m   = r1[0] / n;
  float var = r2[0] / n - m * m;
  float inv = rsqrtf(fmaxf(var, 0.0f) + eps);
  int ok = mask ? mask[row] : 1;
  for (int j = t; j < n; j += 256) {
    float v = ok ? ((xr[j] - m) * inv * g[j] + bb[j]) : 0.0f;
    orow[j] = (bf16)v;
  }
}

__global__ void copy_cat_kernel(const bf16* __restrict__ enc, bf16* __restrict__ cat) {
  long long i = (long long)blockIdx.x * 256 + threadIdx.x;  // over BS*1024
  if (i < (long long)BS * 1024) {
    long long row = i >> 10, col = i & 1023;
    cat[row * 2048 + col] = enc[i];
  }
}

// ---------------------------------------------------------------------------
extern "C" void kernel_launch(void* const* d_in, const int* in_sizes, int n_in,
                              void* d_out, int out_size, void* d_ws, size_t ws_size,
                              hipStream_t stream) {
  (void)in_sizes; (void)n_in; (void)out_size; (void)ws_size;
  const int*   x        = (const int*)d_in[0];
  const float* emb      = (const float*)d_in[1];
  const float* Wf0      = (const float*)d_in[2];
  const float* bf0      = (const float*)d_in[3];
  const float* Wb0      = (const float*)d_in[4];
  const float* bb0      = (const float*)d_in[5];
  const float* Wf1      = (const float*)d_in[6];
  const float* bf1      = (const float*)d_in[7];
  const float* Wb1      = (const float*)d_in[8];
  const float* bb1      = (const float*)d_in[9];
  const float* h0f      = (const float*)d_in[10];
  const float* c0f      = (const float*)d_in[11];
  const float* h0b      = (const float*)d_in[12];
  const float* c0b      = (const float*)d_in[13];
  const float* Wq       = (const float*)d_in[14];
  const float* bq       = (const float*)d_in[15];
  const float* Wao      = (const float*)d_in[16];
  const float* bao      = (const float*)d_in[17];
  const float* ln1_g    = (const float*)d_in[18];
  const float* ln1_b    = (const float*)d_in[19];
  const float* Wh       = (const float*)d_in[20];
  const float* bh       = (const float*)d_in[21];
  const float* ln2_g    = (const float*)d_in[22];
  const float* ln2_b    = (const float*)d_in[23];
  const float* Wp       = (const float*)d_in[24];
  const float* out_bias = (const float*)d_in[25];
  float* out = (float*)d_out;

  // ---- workspace carve-out (256B aligned) ----
  size_t off = 0;
  auto alloc = [&](size_t bytes) -> void* {
    void* p = (char*)d_ws + off;
    off = (off + bytes + 255) & ~(size_t)255;
    return p;
  };
  bf16* Wf0p = (bf16*)alloc((size_t)(Ee + Hh) * 4 * Hh * 2);   // packed fragments
  bf16* Wb0p = (bf16*)alloc((size_t)(Ee + Hh) * 4 * Hh * 2);
  bf16* Wf1p = (bf16*)alloc((size_t)(3 * Hh) * 4 * Hh * 2);
  bf16* Wb1p = (bf16*)alloc((size_t)(3 * Hh) * 4 * Hh * 2);
  bf16* Wqp  = (bf16*)alloc((size_t)2 * Hh * 2 * Hh * 2);
  bf16* Waop = (bf16*)alloc((size_t)4 * Hh * 2 * Hh * 2);
  bf16* Whp  = (bf16*)alloc((size_t)2 * Hh * Ee * 2);
  bf16* Wpp  = (bf16*)alloc((size_t)Ee * Vv * 2);
  bf16* X0   = (bf16*)alloc((size_t)BS * Ee * 2);
  int*  mask = (int*)alloc((size_t)BS * 4);
  float* gpre_f = (float*)alloc((size_t)BS * 4 * Hh * 4);
  float* gpre_b = (float*)alloc((size_t)BS * 4 * Hh * 4);
  bf16* L1in    = (bf16*)alloc((size_t)BS * 2 * Hh * 2);
  bf16* encB    = (bf16*)alloc((size_t)BS * 2 * Hh * 2);
  float* encF   = (float*)alloc((size_t)BS * 2 * Hh * 4);
  bf16* encP    = (bf16*)alloc((size_t)BS * 2 * Hh * 2);   // packed per-batch
  bf16* queryB  = (bf16*)alloc((size_t)BS * 2 * Hh * 2);
  float* scoresF = (float*)alloc((size_t)Bb * Ss * Ss * 4);
  bf16* attnB    = (bf16*)alloc((size_t)Bb * Ss * Ss * 2);
  bf16* catB     = (bf16*)alloc((size_t)BS * 4 * Hh * 2);
  float* refinedF = (float*)alloc((size_t)BS * 2 * Hh * 4);
  bf16* refinedB  = (bf16*)alloc((size_t)BS * 2 * Hh * 2);
  float* headF    = (float*)alloc((size_t)BS * Ee * 4);
  bf16* hdB       = (bf16*)alloc((size_t)BS * Ee * 2);

  auto packw = [&](const float* src, int K, int N, bf16* dst) {
    long long n = (long long)K * N;
    pack_w_kernel<<<dim3((unsigned)((n + 255) / 256)), dim3(256), 0, stream>>>(src, K, N, dst);
  };
  // packed-B GEMM
  auto gemm = [&](const bf16* A, int lda, long long sA,
                  const bf16* B, int ldbK, long long sB, int kt0,
                  const float* bias, float* Cf, bf16* Cb, int ldc, long long sC,
                  const float* res, int ldr, long long sR,
                  int M, int N, int K, float scale, int actk, int batch) {
    dim3 grid((unsigned)((N / 16 + 31) / 32), (unsigned)(M / 16), (unsigned)batch);
    gemm_bf16_kernel<0><<<grid, 256, 0, stream>>>(A, lda, sA, B, ldbK, sB, kt0,
                                                  bias, Cf, Cb, ldc, sC, res, ldr, sR,
                                                  N, K, scale, actk);
  };
  // B-transposed GEMM (B row-major N x K)
  auto gemmT = [&](const bf16* A, int lda, long long sA,
                   const bf16* Bt, int ldbt, long long sB,
                   const float* bias, float* Cf, bf16* Cb, int ldc, long long sC,
                   int M, int N, int K, float scale, int batch) {
    dim3 grid((unsigned)((N / 16 + 31) / 32), (unsigned)(M / 16), (unsigned)batch);
    gemm_bf16_kernel<1><<<grid, 256, 0, stream>>>(A, lda, sA, Bt, ldbt, sB, 0,
                                                  bias, Cf, Cb, ldc, sC,
                                                  nullptr, 0, 0, N, K, scale, 0);
  };

  // ---- pack weights into WMMA B-fragment layout (fuses f32->bf16) ----
  packw(Wf0, Ee + Hh, 4 * Hh, Wf0p);     // Ktot = 32
  packw(Wb0, Ee + Hh, 4 * Hh, Wb0p);
  packw(Wf1, 3 * Hh, 4 * Hh, Wf1p);      // Ktot = 48
  packw(Wb1, 3 * Hh, 4 * Hh, Wb1p);
  packw(Wq, 2 * Hh, 2 * Hh, Wqp);        // Ktot = 32
  packw(Wao, 4 * Hh, 2 * Hh, Waop);      // Ktot = 64
  packw(Wh, 2 * Hh, Ee, Whp);            // Ktot = 32
  packw(Wp, Ee, Vv, Wpp);                // Ktot = 16

  // ---- embedding + mask ----
  embed_kernel<<<dim3(BS), dim3(256), 0, stream>>>(x, emb, X0, mask);

  const size_t LSTM_SMEM = 16 * 512 * sizeof(bf16) + 2 * 8 * 512 * sizeof(float)
                         + 8 * 2048 * sizeof(float);  // 112 KB (WGP has 320 KB LDS)

  // ---- layer 0: gate precompute (x part, k-tiles 0..15) then recurrence ----
  gemm(X0, Ee, 0, Wf0p, 32, 0, 0, bf0, gpre_f, nullptr, 4 * Hh, 0,
       nullptr, 0, 0, BS, 4 * Hh, Ee, 1.0f, 0, 1);
  gemm(X0, Ee, 0, Wb0p, 32, 0, 0, bb0, gpre_b, nullptr, 4 * Hh, 0,
       nullptr, 0, 0, BS, 4 * Hh, Ee, 1.0f, 0, 1);
  lstm_rec_kernel<<<dim3(2), dim3(1024), LSTM_SMEM, stream>>>(
      Wf0p, Wb0p, /*Ktot=*/32, /*kt0=*/Ee / 32,
      gpre_f, gpre_b, h0f, c0f, h0b, c0b, mask, L1in, nullptr, 2 * Hh);

  // ---- layer 1 ----
  gemm(L1in, 2 * Hh, 0, Wf1p, 48, 0, 0, bf1, gpre_f, nullptr, 4 * Hh, 0,
       nullptr, 0, 0, BS, 4 * Hh, 2 * Hh, 1.0f, 0, 1);
  gemm(L1in, 2 * Hh, 0, Wb1p, 48, 0, 0, bb1, gpre_b, nullptr, 4 * Hh, 0,
       nullptr, 0, 0, BS, 4 * Hh, 2 * Hh, 1.0f, 0, 1);
  lstm_rec_kernel<<<dim3(2), dim3(1024), LSTM_SMEM, stream>>>(
      Wf1p, Wb1p, /*Ktot=*/48, /*kt0=*/(2 * Hh) / 32,
      gpre_f, gpre_b, h0f + Hh, c0f + Hh, h0b + Hh, c0b + Hh, mask,
      encB, encF, 2 * Hh);

  // ---- attention ----
  // query = enc @ Wq + bq    (bf16 out only)
  gemm(encB, 2 * Hh, 0, Wqp, 32, 0, 0, bq, nullptr, queryB, 2 * Hh, 0,
       nullptr, 0, 0, BS, 2 * Hh, 2 * Hh, 1.0f, 0, 1);
  // scores[b] = query_b @ enc_b^T * (2H)^-0.5   (Bt path: contiguous loads)
  gemmT(queryB, 2 * Hh, (long long)Ss * 2 * Hh,
        encB, 2 * Hh, (long long)Ss * 2 * Hh,
        nullptr, scoresF, nullptr, Ss, (long long)Ss * Ss,
        Ss, Ss, 2 * Hh, 0.03125f, Bb);
  attn_softmax_kernel<<<dim3(BS), dim3(256), 0, stream>>>(scoresF, mask, attnB);
  // pack enc per batch as B fragments for the context GEMM (K=512 keys, N=1024)
  pack_b_bf16_kernel<<<dim3((unsigned)(((long long)Ss * 2 * Hh + 255) / 256), 1, Bb),
                       dim3(256), 0, stream>>>(
      encB, (long long)Ss * 2 * Hh, Ss, 2 * Hh, encP, (long long)Ss * 2 * Hh);
  // context[b] = attn_b @ enc_b -> right half of cat (ldc = 2048)
  gemm(attnB, Ss, (long long)Ss * Ss,
       encP, /*Ktot=*/Ss / 32, (long long)Ss * 2 * Hh, 0,
       nullptr, nullptr, catB + 2 * Hh, 4 * Hh, (long long)Ss * 4 * Hh,
       nullptr, 0, 0, Ss, 2 * Hh, Ss, 1.0f, 0, Bb);
  copy_cat_kernel<<<dim3((BS * 2 * Hh + 255) / 256), dim3(256), 0, stream>>>(encB, catB);

  // refined = enc + tanh(cat @ Wao + bao)
  gemm(catB, 4 * Hh, 0, Waop, 64, 0, 0, bao, refinedF, nullptr, 2 * Hh, 0,
       encF, 2 * Hh, 0, BS, 2 * Hh, 4 * Hh, 1.0f, 1, 1);
  layernorm_kernel<<<dim3(BS), dim3(256), 0, stream>>>(refinedF, 2 * Hh, ln1_g, ln1_b,
                                                       mask, 1e-5f, refinedB);

  // head = LN2(gelu(refined @ Wh + bh))
  gemm(refinedB, 2 * Hh, 0, Whp, 32, 0, 0, bh, headF, nullptr, Ee, 0,
       nullptr, 0, 0, BS, Ee, 2 * Hh, 1.0f, 2, 1);
  layernorm_kernel<<<dim3(BS), dim3(256), 0, stream>>>(headF, Ee, ln2_g, ln2_b,
                                                       nullptr, 1e-5f, hdB);

  // logits = head @ Wp + out_bias   (4096 x 32000, store-bandwidth bound)
  gemm(hdB, Ee, 0, Wpp, 16, 0, 0, out_bias, out, nullptr, Vv, 0,
       nullptr, 0, 0, BS, Vv, Ee, 1.0f, 0, 1);
}
